// SlotAttention_39075612459641
// MI455X (gfx1250) — compile-verified, matrix operand order checked by reading, not run on hardware
//
#include <hip/hip_runtime.h>
#include <hip/hip_bf16.h>

// ---------------------------------------------------------------------------
// Types for CDNA5 WMMA (wave32) + TDM descriptors
// ---------------------------------------------------------------------------
typedef __attribute__((ext_vector_type(16))) __bf16 v16bf;
typedef __attribute__((ext_vector_type(8)))  float  v8f;
typedef __attribute__((ext_vector_type(4)))  unsigned int v4u;
typedef __attribute__((ext_vector_type(4)))  int    v4i;
typedef __attribute__((ext_vector_type(8)))  int    v8i;

#if defined(__has_builtin)
#if __has_builtin(__builtin_amdgcn_tensor_load_to_lds)
#define USE_TDM_BUILTIN 1
#endif
#endif

union FragU {
    v16bf v;
    float4 f4[2];
};

// Load a 16-lane-striped bf16 WMMA fragment: 8 contiguous halves at p,
// 8 contiguous halves at p+16 (matches 16-bit A/B VGPR layout for 16x16x32).
__device__ __forceinline__ v16bf load_frag32(const __bf16* p) {
    FragU u;
    u.f4[0] = *reinterpret_cast<const float4*>(p);
    u.f4[1] = *reinterpret_cast<const float4*>(p + 16);
    return u.v;
}

__device__ __forceinline__ v8f wmma_bf16(v16bf a, v16bf b, v8f c) {
    return __builtin_amdgcn_wmma_f32_16x16x32_bf16(false, a, false, b, (short)0, c,
                                                   false, false);
}

// ---------------------------------------------------------------------------
// TDM-staged WMMA GEMM: C[M,N] = A[M,K]_bf16 * W[N,K]_bf16^T  (+bias/relu/alpha)
// Block = 128 threads (4 waves). Block tile = 16 rows x 256 cols
// (each wave owns 4 x 16x16 output tiles -> 4 accumulators, A-frag reuse x4).
// The 16xK bf16 A tile (contiguous in memory) is DMA'd to LDS by the Tensor
// Data Mover (tensor_load_to_lds, TENSORcnt), then read back as WMMA A
// fragments via ds_load_b128. W streams from global (B columns contiguous).
// OMODE: 0=store f32, 1=store bf16, 2=accumulate into f32, 3=store bf16
//        transposed per 1024-row batch (vT production).
// Requirements: M%16==0, N%256==0, K%32==0. Dynamic LDS = 32*K bytes.
// ---------------------------------------------------------------------------
template<bool RELU, int OMODE>
__global__ __launch_bounds__(128) void gemm_wmma_tdm_kernel(
    const __bf16* __restrict__ A, const __bf16* __restrict__ W,
    void* __restrict__ C, const float* __restrict__ bias,
    int M, int N, int K, float alpha,
    long long sA, long long sW, long long sC)
{
    extern __shared__ char smem_raw[];
    __bf16* smemA = (__bf16*)smem_raw;

    const int lane = threadIdx.x & 31;
    const int wave = threadIdx.x >> 5;
    const int mt   = blockIdx.x;
    const int z    = blockIdx.z;

    const __bf16* Ab = A + (size_t)z * sA + (size_t)mt * 16 * K;  // tile start
    const __bf16* Wb = W + (size_t)z * sW;

    // ---- stage A tile (16 x K bf16) into LDS ----
#if defined(USE_TDM_BUILTIN)
    if (threadIdx.x < 32) {
        const unsigned long long ga = (unsigned long long)(uintptr_t)Ab;
        v4u g0;
        g0[0] = 1u;                                        // count=1 (valid D#)
        g0[1] = 0u;                                        // lds_addr = dyn-LDS base
        g0[2] = (unsigned)(ga & 0xFFFFFFFFull);            // global_addr[31:0]
        g0[3] = (unsigned)((ga >> 32) & 0x1FFFFFFull) | (2u << 30); // addr hi | type=2
        v8i g1;
        g1[0] = 1 << 16;                                   // data_size = 2 bytes
        g1[1] = (K & 0xFFFF) << 16;                        // tensor_dim0[15:0]
        g1[2] = (K >> 16) | (16 << 16);                    // dim0 hi | tensor_dim1=16
        g1[3] = K << 16;                                   // dim1 hi(0) | tile_dim0=K
        g1[4] = 16;                                        // tile_dim1=16, tile_dim2=0
        g1[5] = K;                                         // tensor_dim0_stride[31:0]
        g1[6] = 0;                                         // stride hi | dim1_stride lo
        g1[7] = 0;
        v4i g2 = {0, 0, 0, 0};
        v4i g3 = {0, 0, 0, 0};
        v8i g4 = {0, 0, 0, 0, 0, 0, 0, 0};
        __builtin_amdgcn_tensor_load_to_lds(g0, g1, g2, g3, g4, 0);
        __builtin_amdgcn_s_wait_tensorcnt(0);              // TENSORcnt -> 0
    }
    __syncthreads();
#else
    {
        const int chunks = (16 * K) >> 3;                  // float4 (8 halves) chunks
        const float4* src = reinterpret_cast<const float4*>(Ab);
        float4* dst = reinterpret_cast<float4*>(smemA);
        for (int i = threadIdx.x; i < chunks; i += 128) dst[i] = src[i];
        __syncthreads();
    }
#endif

    // ---- compute: 4 N-tiles per wave, A frag shared from LDS ----
    const int nt0 = blockIdx.y * 16 + wave * 4;            // first tile (tile units)
    const int ko  = (lane >> 4) << 3;
    const __bf16* asrc = smemA + (size_t)(lane & 15) * K + ko;
    const __bf16* wp0 = Wb + (size_t)((nt0 + 0) * 16 + (lane & 15)) * K + ko;
    const __bf16* wp1 = Wb + (size_t)((nt0 + 1) * 16 + (lane & 15)) * K + ko;
    const __bf16* wp2 = Wb + (size_t)((nt0 + 2) * 16 + (lane & 15)) * K + ko;
    const __bf16* wp3 = Wb + (size_t)((nt0 + 3) * 16 + (lane & 15)) * K + ko;

    v8f acc0 = {0.f,0.f,0.f,0.f,0.f,0.f,0.f,0.f};
    v8f acc1 = acc0, acc2 = acc0, acc3 = acc0;

    for (int k0 = 0; k0 < K; k0 += 32) {
        __builtin_prefetch(wp0 + k0 + 128, 0, 1);          // global_prefetch_b8
        __builtin_prefetch(wp1 + k0 + 128, 0, 1);
        __builtin_prefetch(wp2 + k0 + 128, 0, 1);
        __builtin_prefetch(wp3 + k0 + 128, 0, 1);
        v16bf af = load_frag32(asrc + k0);                 // ds_load_b128 x2
        v16bf w0 = load_frag32(wp0 + k0);
        v16bf w1 = load_frag32(wp1 + k0);
        v16bf w2 = load_frag32(wp2 + k0);
        v16bf w3 = load_frag32(wp3 + k0);
        acc0 = wmma_bf16(af, w0, acc0);
        acc1 = wmma_bf16(af, w1, acc1);
        acc2 = wmma_bf16(af, w2, acc2);
        acc3 = wmma_bf16(af, w3, acc3);
    }

    // ---- epilogue ----
    const int orow0 = mt * 16 + ((lane >> 4) << 3);
    v8f accs[4] = {acc0, acc1, acc2, acc3};
#pragma unroll
    for (int j = 0; j < 4; ++j) {
        const int ocol = (nt0 + j) * 16 + (lane & 15);
        const float bv = bias ? bias[ocol] : 0.f;
#pragma unroll
        for (int g = 0; g < 8; ++g) {
            float v = accs[j][g] * alpha + bv;
            if (RELU) v = v > 0.f ? v : 0.f;
            const int r = orow0 + g;
            if (OMODE == 0) {
                float* p = (float*)C + (size_t)z * sC;
                p[(size_t)r * N + ocol] = v;
            } else if (OMODE == 1) {
                __bf16* p = (__bf16*)C + (size_t)z * sC;
                p[(size_t)r * N + ocol] = (__bf16)v;
            } else if (OMODE == 2) {
                float* p = (float*)C + (size_t)z * sC;
                p[(size_t)r * N + ocol] += v;
            } else {
                __bf16* p = (__bf16*)C;
                const size_t bb = (size_t)(r >> 10);
                const size_t jj = (size_t)(r & 1023);
                p[bb * (size_t)(256 * 1024) + (size_t)ocol * 1024 + jj] = (__bf16)v;
            }
        }
    }
}

// ---------------------------------------------------------------------------
// LayerNorm over rows of 256, writes bf16. PAD: row r=b*8+i -> out row b*16+i
// ---------------------------------------------------------------------------
template<bool PAD>
__global__ __launch_bounds__(256) void ln_kernel(
    const float* __restrict__ in, const float* __restrict__ w,
    const float* __restrict__ b, __bf16* __restrict__ out)
{
    __shared__ float red[256];
    const int r = blockIdx.x, t = threadIdx.x;
    const float x = in[(size_t)r * 256 + t];
    red[t] = x; __syncthreads();
    for (int s = 128; s > 0; s >>= 1) { if (t < s) red[t] += red[t + s]; __syncthreads(); }
    const float mu = red[0] * (1.f / 256.f);
    __syncthreads();
    const float d = x - mu;
    red[t] = d * d; __syncthreads();
    for (int s = 128; s > 0; s >>= 1) { if (t < s) red[t] += red[t + s]; __syncthreads(); }
    const float rs = rsqrtf(red[0] * (1.f / 256.f) + 1e-5f);
    const float y = d * rs * w[t] + b[t];
    const int orow = PAD ? (((r >> 3) << 4) + (r & 7)) : r;
    out[(size_t)orow * 256 + t] = (__bf16)y;
}

__global__ void init_slots_kernel(const float* __restrict__ noise,
                                  const float* __restrict__ mu,
                                  const float* __restrict__ sigma,
                                  float* __restrict__ slots)
{
    const int idx = blockIdx.x * 256 + threadIdx.x;
    const int d = idx & 255;
    slots[idx] = mu[d] + noise[idx] * __expf(sigma[d]);
}

__global__ void cvt_kernel(const float* __restrict__ in, __bf16* __restrict__ out, int n)
{
    const int i = blockIdx.x * 256 + threadIdx.x;
    if (i < n) out[i] = (__bf16)in[i];
}

// updates f32 [64][16][256] -> bf16 [512][256] (strip pad rows)
__global__ void cvt_unpad_kernel(const float* __restrict__ in, __bf16* __restrict__ out)
{
    const int idx = blockIdx.x * 256 + threadIdx.x;
    const int m = idx >> 8, d = idx & 255;
    const int bb = m >> 3, i = m & 7;
    out[idx] = (__bf16)in[(((size_t)bb * 16 + i) << 8) + d];
}

__global__ void zero_bf16_kernel(__bf16* __restrict__ p, int n)
{
    const int i = blockIdx.x * 256 + threadIdx.x;
    if (i < n) p[i] = (__bf16)0.f;
}

// softmax over the 8 slots for each (b, j); attn = softmax + EPS.
__global__ void softmax_kernel(const float* __restrict__ dots,
                               float* __restrict__ attnf,
                               float* __restrict__ attns_out, int it)
{
    const int idx = blockIdx.x * 256 + threadIdx.x;     // 64*1024
    const int b = idx >> 10, j = idx & 1023;
    const float* dp = dots + ((size_t)b << 14) + j;     // [b][16][1024], rows 0..7 valid
    float v[8], m = -3.4e38f;
#pragma unroll
    for (int i = 0; i < 8; ++i) { v[i] = dp[(size_t)i << 10]; m = fmaxf(m, v[i]); }
    float s = 0.f;
#pragma unroll
    for (int i = 0; i < 8; ++i) { v[i] = __expf(v[i] - m); s += v[i]; }
    const float inv = 1.f / s;
#pragma unroll
    for (int i = 0; i < 8; ++i) {
        const float a = v[i] * inv + 1e-8f;
        attnf[(((size_t)b * 8 + i) << 10) + j] = a;
        attns_out[((((size_t)b * 3 + it) * 8 + i) << 10) + j] = a;
    }
}

// attn_n = attn / sum_j attn ; write bf16 into padded [b][16][1024] A-matrix.
__global__ __launch_bounds__(256) void attn_norm_kernel(
    const float* __restrict__ attnf, __bf16* __restrict__ attnn)
{
    __shared__ float red[256];
    const int bi = blockIdx.x;                 // b*8+i  (512)
    const int t = threadIdx.x;
    const float* p = attnf + ((size_t)bi << 10);
    float s = p[t] + p[t + 256] + p[t + 512] + p[t + 768];
    red[t] = s; __syncthreads();
    for (int st = 128; st > 0; st >>= 1) { if (t < st) red[t] += red[t + st]; __syncthreads(); }
    const float inv = 1.f / red[0];
    __bf16* o = attnn + ((((size_t)(bi >> 3) << 4) + (bi & 7)) << 10);
#pragma unroll
    for (int q = 0; q < 4; ++q) o[t + q * 256] = (__bf16)(p[t + q * 256] * inv);
}

// GRU combine: gi,gh [512][768], slots in/out [512][256]
__global__ void gru_combine_kernel(const float* __restrict__ gi,
                                   const float* __restrict__ gh,
                                   float* __restrict__ slots)
{
    const int idx = blockIdx.x * 256 + threadIdx.x;
    const int m = idx >> 8, c = idx & 255;
    const size_t base = (size_t)m * 768;
    const float ir = gi[base + c],        hr = gh[base + c];
    const float iz = gi[base + 256 + c],  hz = gh[base + 256 + c];
    const float in_ = gi[base + 512 + c], hn = gh[base + 512 + c];
    const float r = 1.f / (1.f + __expf(-(ir + hr)));
    const float z = 1.f / (1.f + __expf(-(iz + hz)));
    const float ng = tanhf(in_ + r * hn);
    const float h = slots[idx];
    slots[idx] = (1.f - z) * ng + z * h;
}

__global__ void copy_f32_kernel(const float* __restrict__ in, float* __restrict__ out, int n)
{
    const int i = blockIdx.x * 256 + threadIdx.x;
    if (i < n) out[i] = in[i];
}

// ---------------------------------------------------------------------------
// Host launcher
// ---------------------------------------------------------------------------
extern "C" void kernel_launch(void* const* d_in, const int* in_sizes, int n_in,
                              void* d_out, int out_size, void* d_ws, size_t ws_size,
                              hipStream_t stream)
{
    const float* inputs  = (const float*)d_in[0];
    const float* noise   = (const float*)d_in[1];
    const float* mu      = (const float*)d_in[2];
    const float* sigma   = (const float*)d_in[3];
    const float* Wq      = (const float*)d_in[4];
    const float* Wk      = (const float*)d_in[5];
    const float* Wv      = (const float*)d_in[6];
    const float* wih     = (const float*)d_in[7];
    const float* whh     = (const float*)d_in[8];
    const float* bih     = (const float*)d_in[9];
    const float* bhh     = (const float*)d_in[10];
    const float* w1      = (const float*)d_in[11];
    const float* b1      = (const float*)d_in[12];
    const float* w2      = (const float*)d_in[13];
    const float* b2      = (const float*)d_in[14];
    const float* ln_in_w = (const float*)d_in[15];
    const float* ln_in_b = (const float*)d_in[16];
    const float* ln_s_w  = (const float*)d_in[17];
    const float* ln_s_b  = (const float*)d_in[18];
    const float* ln_f_w  = (const float*)d_in[19];
    const float* ln_f_b  = (const float*)d_in[20];

    char* ws = (char*)d_ws;
    size_t off = 0;
    auto take = [&](size_t bytes) -> char* {
        char* p = ws + off;
        off += (bytes + 255) & ~(size_t)255;
        return p;
    };

    __bf16* Xb     = (__bf16*)take((size_t)65536 * 256 * 2);   // LN(inputs) bf16
    __bf16* Kb     = (__bf16*)take((size_t)65536 * 256 * 2);   // k [b][1024][256]
    __bf16* VTb    = (__bf16*)take((size_t)65536 * 256 * 2);   // vT [b][256][1024]
    __bf16* Wqb    = (__bf16*)take((size_t)65536 * 2);
    __bf16* Wkb    = (__bf16*)take((size_t)65536 * 2);
    __bf16* Wvb    = (__bf16*)take((size_t)65536 * 2);
    __bf16* Wihb   = (__bf16*)take((size_t)196608 * 2);
    __bf16* Whhb   = (__bf16*)take((size_t)196608 * 2);
    __bf16* W1b    = (__bf16*)take((size_t)131072 * 2);
    __bf16* W2b    = (__bf16*)take((size_t)131072 * 2);
    float*  SlotsF = (float*) take((size_t)131072 * 4);        // [512][256]
    __bf16* SprevB = (__bf16*)take((size_t)131072 * 2);
    __bf16* SpadB  = (__bf16*)take((size_t)262144 * 2);        // [64][16][256]
    __bf16* Qb     = (__bf16*)take((size_t)262144 * 2);        // [64][16][256]
    float*  DotsF  = (float*) take((size_t)64 * 16 * 1024 * 4);
    float*  AttnF  = (float*) take((size_t)64 * 8 * 1024 * 4);
    __bf16* AttnNb = (__bf16*)take((size_t)64 * 16 * 1024 * 2);
    float*  UpdF   = (float*) take((size_t)64 * 16 * 256 * 4);
    __bf16* UpdB   = (__bf16*)take((size_t)131072 * 2);
    float*  GiF    = (float*) take((size_t)512 * 768 * 4);
    float*  GhF    = (float*) take((size_t)512 * 768 * 4);
    __bf16* Tb     = (__bf16*)take((size_t)131072 * 2);
    __bf16* H1b    = (__bf16*)take((size_t)262144 * 2);        // [512][512]

    const float scale = 0.0625f;   // 256^-0.5

    // ----- one-time per call -----
    ln_kernel<false><<<dim3(65536), dim3(256), 0, stream>>>(inputs, ln_in_w, ln_in_b, Xb);
    cvt_kernel<<<dim3(256),  dim3(256), 0, stream>>>(Wq,  Wqb,  65536);
    cvt_kernel<<<dim3(256),  dim3(256), 0, stream>>>(Wk,  Wkb,  65536);
    cvt_kernel<<<dim3(256),  dim3(256), 0, stream>>>(Wv,  Wvb,  65536);
    cvt_kernel<<<dim3(768),  dim3(256), 0, stream>>>(wih, Wihb, 196608);
    cvt_kernel<<<dim3(768),  dim3(256), 0, stream>>>(whh, Whhb, 196608);
    cvt_kernel<<<dim3(512),  dim3(256), 0, stream>>>(w1,  W1b,  131072);
    cvt_kernel<<<dim3(512),  dim3(256), 0, stream>>>(w2,  W2b,  131072);
    init_slots_kernel<<<dim3(512), dim3(256), 0, stream>>>(noise, mu, sigma, SlotsF);
    zero_bf16_kernel<<<dim3(1024), dim3(256), 0, stream>>>(SpadB, 262144);
    zero_bf16_kernel<<<dim3(4096), dim3(256), 0, stream>>>(AttnNb, 64 * 16 * 1024);

    // K = LN(x) @ Wk^T ; V = LN(x) @ Wv^T (stored transposed per batch)
    gemm_wmma_tdm_kernel<false, 1><<<dim3(4096, 1, 1), dim3(128), 32 * 256, stream>>>(
        Xb, Wkb, Kb, nullptr, 65536, 256, 256, 1.f, 0, 0, 0);
    gemm_wmma_tdm_kernel<false, 3><<<dim3(4096, 1, 1), dim3(128), 32 * 256, stream>>>(
        Xb, Wvb, VTb, nullptr, 65536, 256, 256, 1.f, 0, 0, 0);

    float* attns_out = (float*)d_out + 512 * 256;

    for (int it = 0; it < 3; ++it) {
        // q = LN(slots) @ Wq^T   (padded to 16 slot rows)
        ln_kernel<true><<<dim3(512), dim3(256), 0, stream>>>(SlotsF, ln_s_w, ln_s_b, SpadB);
        gemm_wmma_tdm_kernel<false, 1><<<dim3(64, 1, 1), dim3(128), 32 * 256, stream>>>(
            SpadB, Wqb, Qb, nullptr, 1024, 256, 256, 1.f, 0, 0, 0);

        // dots[b] = scale * q[b] @ k[b]^T   (batched: z = 64)
        gemm_wmma_tdm_kernel<false, 0><<<dim3(1, 4, 64), dim3(128), 32 * 256, stream>>>(
            Qb, Kb, DotsF, nullptr, 16, 1024, 256, scale,
            (long long)16 * 256, (long long)1024 * 256, (long long)16 * 1024);

        softmax_kernel<<<dim3(256), dim3(256), 0, stream>>>(DotsF, AttnF, attns_out, it);
        attn_norm_kernel<<<dim3(512), dim3(256), 0, stream>>>(AttnF, AttnNb);

        // updates[b] = attn_n[b] @ v[b]   (B columns contiguous via vT)
        gemm_wmma_tdm_kernel<false, 0><<<dim3(1, 1, 64), dim3(128), 32 * 1024, stream>>>(
            AttnNb, VTb, UpdF, nullptr, 16, 256, 1024, 1.f,
            (long long)16 * 1024, (long long)256 * 1024, (long long)16 * 256);

        cvt_unpad_kernel<<<dim3(512), dim3(256), 0, stream>>>(UpdF, UpdB);
        cvt_kernel<<<dim3(512), dim3(256), 0, stream>>>(SlotsF, SprevB, 131072);

        // GRU gates
        gemm_wmma_tdm_kernel<false, 0><<<dim3(32, 3, 1), dim3(128), 32 * 256, stream>>>(
            UpdB, Wihb, GiF, bih, 512, 768, 256, 1.f, 0, 0, 0);
        gemm_wmma_tdm_kernel<false, 0><<<dim3(32, 3, 1), dim3(128), 32 * 256, stream>>>(
            SprevB, Whhb, GhF, bhh, 512, 768, 256, 1.f, 0, 0, 0);
        gru_combine_kernel<<<dim3(512), dim3(256), 0, stream>>>(GiF, GhF, SlotsF);

        // MLP residual
        ln_kernel<false><<<dim3(512), dim3(256), 0, stream>>>(SlotsF, ln_f_w, ln_f_b, Tb);
        gemm_wmma_tdm_kernel<true, 1><<<dim3(32, 2, 1), dim3(128), 32 * 256, stream>>>(
            Tb, W1b, H1b, b1, 512, 512, 256, 1.f, 0, 0, 0);
        gemm_wmma_tdm_kernel<false, 2><<<dim3(32, 1, 1), dim3(128), 32 * 512, stream>>>(
            H1b, W2b, SlotsF, b2, 512, 256, 512, 1.f, 0, 0, 0);
    }

    copy_f32_kernel<<<dim3(512), dim3(256), 0, stream>>>(SlotsF, (float*)d_out, 131072);
}
